// MyGNN_65171833749504
// MI455X (gfx1250) — compile-verified, matrix operand order checked by reading, not run on hardware
//
#include <hip/hip_runtime.h>

#define XS   136   // LDS row stride in halfs for x/agg (16B-aligned rows, bank-skewed)
#define FS   72    // LDS row stride in halfs for encoder features
#define GPB  16    // graphs per block -> 208 rows = 13 WMMA m-tiles
#define ROWS 208

typedef __attribute__((ext_vector_type(16))) _Float16 v16h;
typedef __attribute__((ext_vector_type(8)))  _Float16 v8h;
typedef __attribute__((ext_vector_type(8)))  float    v8f;

// packed tree adjacency: bits[3:0]=cnt, then 4-bit neighbor ids
__constant__ unsigned kNbr[13] = {
  0xA7414u, 0x202u, 0x312u, 0x21u, 0x502u, 0x642u, 0x51u,
  0x802u,   0x972u, 0x81u,  0xB02u, 0xCA2u, 0xB1u
};

// A-fragment (16x32 f16, M across lanes 0..15, K split per CDNA5 layout):
// lane reads halfs [k .. k+7] and [k+16 .. k+23] of its row -> 2x ds_load_b128
__device__ inline v16h ldsA(const _Float16* p) {
  v8h lo = *(const v8h*)p;
  v8h hi = *(const v8h*)(p + 16);
  v16h r;
#pragma unroll
  for (int i = 0; i < 8; ++i) { r[i] = lo[i]; r[i + 8] = hi[i]; }
  return r;
}

// B-fragment from a row-major fp32 weight row (out = x @ W^T => lane n holds W row n)
__device__ inline v16h gblB(const float* p) {
  v16h r;
#pragma unroll
  for (int i = 0; i < 8; ++i) { r[i] = (_Float16)p[i]; r[i + 8] = (_Float16)p[16 + i]; }
  return r;
}

__device__ inline float elu1(float f) { return f > 0.f ? f : (__expf(f) - 1.f); }

__global__ __launch_bounds__(256)
void gnn_fused(const float* __restrict__ obs,
               const float* __restrict__ We_b, const float* __restrict__ be_b,
               const float* __restrict__ We_j, const float* __restrict__ be_j,
               const float* __restrict__ W_rel, const float* __restrict__ W_root,
               const float* __restrict__ b_rel,
               const float* __restrict__ W_dec, const float* __restrict__ b_dec,
               float* __restrict__ out)
{
  __shared__ __align__(16) _Float16 sX[ROWS * XS];   // node state (f16)
  __shared__ __align__(16) _Float16 sG[ROWS * XS];   // neighbor aggregate / features

  const int tid  = threadIdx.x;
  const int blk  = blockIdx.x;
  const int lane = tid & 31;
  const int wave = tid >> 5;        // 0..7 -> N-tile of H
  const int lm   = lane & 15;
  const int hi   = lane >> 4;
  const int koff = hi * 8;          // K offset for upper half-wave (CDNA5 16-bit A layout)
  const int ncol = wave * 16 + lm;  // output channel this lane produces

  // ---------- stage obs rows for 16 graphs (reuse sX as fp32 scratch) ----------
  float* sObs = (float*)sX;                      // 16 x 144 fp32
  for (int i = tid; i < GPB * 141; i += 256) {
    int g = i / 141, w = i - g * 141;
    sObs[g * 144 + w] = obs[(blk * GPB + g) * 141 + w];
  }
  __syncthreads();

  // ---------- assemble unified per-node features into sG: [208 x 64] f16 ----------
  // cols 0..32: base features (node 0), cols 33..41: joint features, rest 0
  _Float16* sF = sG;
  for (int idx = tid; idx < ROWS * 8; idx += 256) {
    int r = idx >> 3, seg = idx & 7;
    int g = r / 13, n = r - g * 13;
    const float* ob = sObs + g * 144;
    _Float16* fr = sF + r * FS + seg * 8;
#pragma unroll
    for (int c = 0; c < 8; ++c) {
      int f = seg * 8 + c;
      float v = 0.f;
      if (n == 0) {
        if (f < 33) {
          int t = f / 11, j = f - t * 11;
          int col = (j < 9) ? j : (36 + j);     // BASE_IDX: 0..8,45,46
          v = ob[t * 47 + col];
        }
      } else if (f >= 33 && f < 42) {
        int q = f - 33;
        int t = q / 3, cc = q - t * 3;
        v = ob[t * 47 + 9 + cc * 12 + (n - 1)]; // JOINT_IDX layout
      }
      fr[c] = (_Float16)v;
    }
  }
  __syncthreads();

  _Float16 keep[13][8];   // layer outputs held in regs between barriers

  // ---------- encoder: x = elu(feat @ Wcomb^T + bias)  (K=64 -> 2 k-tiles) ----------
  {
    v16h wb[2];
#pragma unroll
    for (int kt = 0; kt < 2; ++kt) {
#pragma unroll
      for (int i = 0; i < 8; ++i) {
        int k0 = kt * 32 + koff + i;
        int k1 = k0 + 16;
        float v0 = (k0 < 33) ? We_b[ncol * 33 + k0]
                 : ((k0 < 42) ? We_j[ncol * 9 + (k0 - 33)] : 0.f);
        float v1 = (k1 < 33) ? We_b[ncol * 33 + k1]
                 : ((k1 < 42) ? We_j[ncol * 9 + (k1 - 33)] : 0.f);
        wb[kt][i]     = (_Float16)v0;
        wb[kt][i + 8] = (_Float16)v1;
      }
    }
    float bb = be_b[ncol], bj = be_j[ncol];
#pragma unroll
    for (int mt = 0; mt < 13; ++mt) {
      v8f acc;
#pragma unroll
      for (int v = 0; v < 8; ++v) {
        int r = mt * 16 + hi * 8 + v;           // D-matrix row for this VGPR
        acc[v] = ((r % 13) == 0) ? bb : bj;
      }
      const _Float16* ap = sF + (mt * 16 + lm) * FS + koff;
#pragma unroll
      for (int kt = 0; kt < 2; ++kt) {
        v16h a = ldsA(ap + kt * 32);
        acc = __builtin_amdgcn_wmma_f32_16x16x32_f16(false, a, false, wb[kt],
                                                     (short)0, acc, false, false);
      }
#pragma unroll
      for (int v = 0; v < 8; ++v) keep[mt][v] = (_Float16)elu1(acc[v]);
    }
  }
  __syncthreads();
#pragma unroll
  for (int mt = 0; mt < 13; ++mt)
#pragma unroll
    for (int v = 0; v < 8; ++v)
      sX[(mt * 16 + hi * 8 + v) * XS + ncol] = keep[mt][v];
  __syncthreads();

  // ---------- 3 GraphConv layers ----------
  for (int l = 0; l < 3; ++l) {
    // fixed-tree scatter-add: sG[r] = sum over neighbors of sX (v_pk_add_f16)
    for (int idx = tid; idx < ROWS * 16; idx += 256) {
      int r = idx >> 4, seg = idx & 15;
      int g = r / 13, n = r - g * 13;
      unsigned pk = kNbr[n];
      int cnt = pk & 0xF;
      v8h s = {};
#pragma unroll
      for (int j = 0; j < 4; ++j) {
        if (j < cnt) {
          int nb = (pk >> (4 + 4 * j)) & 0xF;
          s += *(const v8h*)(sX + (g * 13 + nb) * XS + seg * 8);
        }
      }
      *(v8h*)(sG + r * XS + seg * 8) = s;
    }
    __syncthreads();

    // per-wave loop-invariant weight fragments (64 VGPRs)
    const float* Wrel  = W_rel  + l * 128 * 128 + ncol * 128 + koff;
    const float* Wroot = W_root + l * 128 * 128 + ncol * 128 + koff;
    v16h brel[4], broot[4];
#pragma unroll
    for (int kt = 0; kt < 4; ++kt) {
      brel[kt]  = gblB(Wrel  + kt * 32);
      broot[kt] = gblB(Wroot + kt * 32);
    }
    float bias = b_rel[l * 128 + ncol];

#pragma unroll
    for (int mt = 0; mt < 13; ++mt) {
      v8f acc;
#pragma unroll
      for (int v = 0; v < 8; ++v) acc[v] = bias;
      const _Float16* ag = sG + (mt * 16 + lm) * XS + koff;
      const _Float16* ax = sX + (mt * 16 + lm) * XS + koff;
#pragma unroll
      for (int kt = 0; kt < 4; ++kt) {
        v16h a0 = ldsA(ag + kt * 32);
        acc = __builtin_amdgcn_wmma_f32_16x16x32_f16(false, a0, false, brel[kt],
                                                     (short)0, acc, false, false);
        v16h a1 = ldsA(ax + kt * 32);
        acc = __builtin_amdgcn_wmma_f32_16x16x32_f16(false, a1, false, broot[kt],
                                                     (short)0, acc, false, false);
      }
#pragma unroll
      for (int v = 0; v < 8; ++v) keep[mt][v] = (_Float16)elu1(acc[v]);
    }
    __syncthreads();
#pragma unroll
    for (int mt = 0; mt < 13; ++mt)
#pragma unroll
      for (int v = 0; v < 8; ++v)
        sX[(mt * 16 + hi * 8 + v) * XS + ncol] = keep[mt][v];
    __syncthreads();
  }

  // ---------- per-joint decoder: out[g, j] = x[g,1+j] . W_dec + b_dec ----------
  if (tid < 192) {
    int g = tid / 12, j = tid - g * 12;
    const _Float16* xr = sX + (g * 13 + 1 + j) * XS;
    float s = b_dec[0];
#pragma unroll
    for (int h8 = 0; h8 < 16; ++h8) {
      v8h xv = *(const v8h*)(xr + h8 * 8);
#pragma unroll
      for (int i = 0; i < 8; ++i) s += (float)xv[i] * W_dec[h8 * 8 + i];
    }
    out[blk * 192 + tid] = s;   // contiguous, coalesced
  }
}

extern "C" void kernel_launch(void* const* d_in, const int* in_sizes, int n_in,
                              void* d_out, int out_size, void* d_ws, size_t ws_size,
                              hipStream_t stream) {
  const float* obs    = (const float*)d_in[0];
  const float* We_b   = (const float*)d_in[1];
  const float* be_b   = (const float*)d_in[2];
  const float* We_j   = (const float*)d_in[3];
  const float* be_j   = (const float*)d_in[4];
  const float* W_rel  = (const float*)d_in[5];
  const float* W_root = (const float*)d_in[6];
  const float* b_rel  = (const float*)d_in[7];
  const float* W_dec  = (const float*)d_in[8];
  const float* b_dec  = (const float*)d_in[9];
  float* out = (float*)d_out;

  int B = in_sizes[0] / 141;          // T*OBS_W = 141 floats per graph
  int blocks = B / GPB;               // 2048
  hipLaunchKernelGGL(gnn_fused, dim3(blocks), dim3(256), 0, stream,
                     obs, We_b, be_b, We_j, be_j, W_rel, W_root, b_rel,
                     W_dec, b_dec, out);
}
